// Prot3DGraphModel1_84430467105442
// MI455X (gfx1250) — matrix operand order repaired and checked by reading, not compiled, not run on hardware
//
#include <hip/hip_runtime.h>
#include <hip/hip_bf16.h>
#include <math.h>

// Problem constants (fixed by the reference)
#define BB   32
#define NPG  70
#define NN   (BB * NPG)     // 2240 nodes
#define EE   (NN * 69)      // 154560 edges

typedef __attribute__((ext_vector_type(16))) __bf16 v16bf;
typedef __attribute__((ext_vector_type(8)))  __bf16 bf16x8;
typedef __attribute__((ext_vector_type(8)))  float  v8f;

// ---------------------------------------------------------------------------
// WMMA bf16 GEMM:  C[M,N] = A[M,Kp] (bf16, zero-padded) * WT[N,Kp]^T + bias
// A row-major [M,Kp]; WT is the weight TRANSPOSED: WT[n*Kp + k] = W[k][n].
// Each wave computes a 64x16 output block (4 M-tiles sharing one B fragment).
// Fragment layouts per ISA 7.12.2 (16-bit, wave32):
//   A 16x32: lane g*16+m, VGPR v: pairs K=(v<4?0:16)+(v&3)*2+g*8
//            -> per lane contiguous K [g*8,g*8+8) and [16+g*8,16+g*8+8)
//   B 32x16: lane g*16+n, same K order as A -> K [g*8,g*8+8) and [16+g*8,16+g*8+8)
//   C/D:     VGPR r -> row (g?8:0)+r, col = lane&15
// ---------------------------------------------------------------------------
__device__ __forceinline__ v16bf combine16(bf16x8 lo, bf16x8 hi) {
  v16bf r;
#pragma unroll
  for (int i = 0; i < 8; ++i) { r[i] = lo[i]; r[8 + i] = hi[i]; }
  return r;
}

__global__ void gemm_wmma_bf16(const __bf16* __restrict__ A, const __bf16* __restrict__ WT,
                               const float* __restrict__ bias, float* __restrict__ C,
                               int M, int Kp, int Nout) {
  const int lane = threadIdx.x & 31;
  const int wave = threadIdx.x >> 5;
  const int wavesPerBlock = blockDim.x >> 5;
  const int tile = blockIdx.x * wavesPerBlock + wave;
  const int tilesN = Nout >> 4;
  const int tilesM = M >> 6;                 // 64-row super-tiles
  if (tile >= tilesM * tilesN) return;       // wave-uniform: EXEC stays all-1s
  const int tm = tile / tilesN;
  const int tn = tile - tm * tilesN;
  const int row0 = tm << 6;
  const int col0 = tn << 4;

  const int g = (lane >> 4) & 1;
  const int ln = lane & 15;

  // per-lane base pointers
  const __bf16* Ar = A + (size_t)(row0 + ln) * Kp + (g << 3);      // +k0, +16
  const __bf16* Wr = WT + (size_t)(col0 + ln) * Kp + (g << 3);     // +k0, +16

  v8f acc0 = {}, acc1 = {}, acc2 = {}, acc3 = {};
  const size_t rowStep = (size_t)16 * Kp;

  for (int k0 = 0; k0 < Kp; k0 += 32) {
    bf16x8 b0 = *(const bf16x8*)(Wr + k0);
    bf16x8 b1 = *(const bf16x8*)(Wr + k0 + 16);
    v16bf bf = combine16(b0, b1);

    v16bf a0 = combine16(*(const bf16x8*)(Ar + k0),
                         *(const bf16x8*)(Ar + k0 + 16));
    v16bf a1 = combine16(*(const bf16x8*)(Ar + rowStep + k0),
                         *(const bf16x8*)(Ar + rowStep + k0 + 16));
    v16bf a2 = combine16(*(const bf16x8*)(Ar + 2 * rowStep + k0),
                         *(const bf16x8*)(Ar + 2 * rowStep + k0 + 16));
    v16bf a3 = combine16(*(const bf16x8*)(Ar + 3 * rowStep + k0),
                         *(const bf16x8*)(Ar + 3 * rowStep + k0 + 16));

    acc0 = __builtin_amdgcn_wmma_f32_16x16x32_bf16(false, a0, false, bf, (short)0, acc0, false, false);
    acc1 = __builtin_amdgcn_wmma_f32_16x16x32_bf16(false, a1, false, bf, (short)0, acc1, false, false);
    acc2 = __builtin_amdgcn_wmma_f32_16x16x32_bf16(false, a2, false, bf, (short)0, acc2, false, false);
    acc3 = __builtin_amdgcn_wmma_f32_16x16x32_bf16(false, a3, false, bf, (short)0, acc3, false, false);
  }

  const int n = col0 + ln;
  const float bv = bias ? bias[n] : 0.0f;
  float* Cp = C + (size_t)(row0 + (g << 3)) * Nout + n;
#pragma unroll
  for (int r = 0; r < 8; ++r) Cp[(size_t)r * Nout] = acc0[r] + bv;
  Cp += (size_t)16 * Nout;
#pragma unroll
  for (int r = 0; r < 8; ++r) Cp[(size_t)r * Nout] = acc1[r] + bv;
  Cp += (size_t)16 * Nout;
#pragma unroll
  for (int r = 0; r < 8; ++r) Cp[(size_t)r * Nout] = acc2[r] + bv;
  Cp += (size_t)16 * Nout;
#pragma unroll
  for (int r = 0; r < 8; ++r) Cp[(size_t)r * Nout] = acc3[r] + bv;
}

// ---------------------------------------------------------------------------
// Conversion / packing kernels (f32 -> bf16, zero-padded K)
// ---------------------------------------------------------------------------
__global__ void convert_pad_kernel(const float* __restrict__ in, __bf16* __restrict__ out,
                                   int M, int K, int Kp) {
  int i = blockIdx.x * blockDim.x + threadIdx.x;       // over M*Kp
  if (i >= M * Kp) return;
  int m = i / Kp, k = i - m * Kp;
  out[i] = (__bf16)((k < K) ? in[(size_t)m * K + k] : 0.0f);
}

// WT[n*Kp + k] = W[k*Nout + n]  (transpose + convert + pad)
__global__ void convert_wt_kernel(const float* __restrict__ W, __bf16* __restrict__ out,
                                  int K, int Kp, int Nout) {
  int i = blockIdx.x * blockDim.x + threadIdx.x;       // over Nout*Kp
  if (i >= Nout * Kp) return;
  int n = i / Kp, k = i - n * Kp;
  out[i] = (__bf16)((k < K) ? W[(size_t)k * Nout + n] : 0.0f);
}

// concat(node_s[.,6], seq[.,20]) -> bf16 [NN, 32] zero-padded
__global__ void concat_convert_kernel(const float* __restrict__ node_s,
                                      const float* __restrict__ seq,
                                      __bf16* __restrict__ out) {
  int i = blockIdx.x * blockDim.x + threadIdx.x;       // over NN*32
  if (i >= NN * 32) return;
  int m = i >> 5, c = i & 31;
  float v = 0.0f;
  if (c < 6)        v = node_s[m * 6 + c];
  else if (c < 26)  v = seq[m * 20 + (c - 6)];
  out[i] = (__bf16)v;
}

// ---------------------------------------------------------------------------
// Elementwise / utility kernels
// ---------------------------------------------------------------------------
__global__ void fill_f32(float* p, float v, int n) {
  int i = blockIdx.x * blockDim.x + threadIdx.x;
  if (i < n) p[i] = v;
}
__global__ void fill_u32(unsigned* p, unsigned v, int n) {
  int i = blockIdx.x * blockDim.x + threadIdx.x;
  if (i < n) p[i] = v;
}

// leaky-relu in place AND emit bf16 copy for the next layer's GEMMs
__global__ void lrelu_convert_kernel(float* __restrict__ x, __bf16* __restrict__ xb, int n) {
  int i = blockIdx.x * blockDim.x + threadIdx.x;
  if (i >= n) return;
  float v = x[i];
  v = (v > 0.0f) ? v : 0.01f * v;
  x[i] = v;
  xb[i] = (__bf16)v;
}

// ordered-uint encoding for float atomic max
__device__ __forceinline__ unsigned enc_f32(float f) {
  unsigned u = __float_as_uint(f);
  return (u & 0x80000000u) ? ~u : (u | 0x80000000u);
}
__device__ __forceinline__ float dec_f32(unsigned k) {
  unsigned u = (k & 0x80000000u) ? (k & 0x7FFFFFFFu) : ~k;
  return __uint_as_float(u);
}

// alpha[e] = dot(q[dst], k[src] + el[e]) * scale ; segment max into mkey[dst]
__global__ void edge_alpha_kernel(const float* __restrict__ q, const float* __restrict__ kk,
                                  const float* __restrict__ el,
                                  const int* __restrict__ src, const int* __restrict__ dst,
                                  float* __restrict__ alpha, unsigned* __restrict__ mkey,
                                  int E, int cout, float scale) {
  int e = blockIdx.x * blockDim.x + threadIdx.x;
  if (e >= E) return;
  int s = src[e], d = dst[e];
  const float4* qd = (const float4*)(q + (size_t)d * cout);
  const float4* ks = (const float4*)(kk + (size_t)s * cout);
  const float4* ee = (const float4*)(el + (size_t)e * cout);
  __builtin_prefetch(el + (size_t)(e + 1) * cout, 0, 1);   // global_prefetch_b8
  float acc = 0.0f;
  const int c4 = cout >> 2;
  for (int c = 0; c < c4; ++c) {
    float4 a = qd[c], b = ks[c], g = ee[c];
    acc += a.x * (b.x + g.x) + a.y * (b.y + g.y) + a.z * (b.z + g.z) + a.w * (b.w + g.w);
  }
  acc *= scale;
  alpha[e] = acc;
  atomicMax(&mkey[d], enc_f32(acc));
}

// aexp[e] = exp(alpha - m[dst]); segment sum into ssum[dst]
__global__ void edge_exp_kernel(const float* __restrict__ alpha,
                                const int* __restrict__ dst,
                                const unsigned* __restrict__ mkey,
                                float* __restrict__ aexp, float* __restrict__ ssum, int E) {
  int e = blockIdx.x * blockDim.x + threadIdx.x;
  if (e >= E) return;
  int d = dst[e];
  float a = __expf(alpha[e] - dec_f32(mkey[d]));
  aexp[e] = a;
  atomicAdd(&ssum[d], a);
}

// out[dst] += (aexp/(ssum+eps)) * (v[src] + el[e])   (out pre-loaded with skip GEMM)
__global__ void edge_msg_kernel(const float* __restrict__ vv, const float* __restrict__ el,
                                const int* __restrict__ src, const int* __restrict__ dst,
                                const float* __restrict__ aexp, const float* __restrict__ ssum,
                                float* __restrict__ out, int E, int cout) {
  int e = blockIdx.x * blockDim.x + threadIdx.x;
  if (e >= E) return;
  int s = src[e], d = dst[e];
  float w = aexp[e] / (ssum[d] + 1e-16f);
  const float4* vs = (const float4*)(vv + (size_t)s * cout);
  const float4* ee = (const float4*)(el + (size_t)e * cout);
  float* od = out + (size_t)d * cout;
  const int c4 = cout >> 2;
  for (int c = 0; c < c4; ++c) {
    float4 a = vs[c], b = ee[c];
    atomicAdd(&od[4 * c + 0], w * (a.x + b.x));
    atomicAdd(&od[4 * c + 1], w * (a.y + b.y));
    atomicAdd(&od[4 * c + 2], w * (a.z + b.z));
    atomicAdd(&od[4 * c + 3], w * (a.w + b.w));
  }
}

__global__ void pool_kernel(const float* __restrict__ x, float* __restrict__ pooled) {
  int i = blockIdx.x * blockDim.x + threadIdx.x;   // over 32*256
  if (i >= BB * 256) return;
  int b = i >> 8, c = i & 255;
  float acc = 0.0f;
  for (int n = 0; n < NPG; ++n) acc += x[(size_t)(b * NPG + n) * 256 + c];
  pooled[i] = acc * (1.0f / (float)NPG);
}

// small dense layer, act: 0=none, 1=relu, 2=sigmoid
__global__ void fc_kernel(const float* __restrict__ A, const float* __restrict__ W,
                          const float* __restrict__ bias, float* __restrict__ C,
                          int M, int K, int Nout, int act) {
  int i = blockIdx.x * blockDim.x + threadIdx.x;
  if (i >= M * Nout) return;
  int m = i / Nout, n = i - m * Nout;
  float acc = bias ? bias[n] : 0.0f;
  const float* Ar = A + (size_t)m * K;
  for (int k = 0; k < K; ++k) acc += Ar[k] * W[(size_t)k * Nout + n];
  if (act == 1) acc = fmaxf(acc, 0.0f);
  else if (act == 2) acc = 1.0f / (1.0f + __expf(-acc));
  C[i] = acc;
}

// ---------------------------------------------------------------------------
// Host driver
// ---------------------------------------------------------------------------
static inline size_t alignUp(size_t v, size_t a) { return (v + a - 1) & ~(a - 1); }

extern "C" void kernel_launch(void* const* d_in, const int* in_sizes, int n_in,
                              void* d_out, int out_size, void* d_ws, size_t ws_size,
                              hipStream_t stream) {
  (void)in_sizes; (void)n_in; (void)out_size; (void)ws_size;

  // ---- inputs (setup_inputs dict order, params flattened depth-first) ----
  const float* node_s = (const float*)d_in[0];
  const float* seq    = (const float*)d_in[1];
  const int*   eidx   = (const int*)d_in[2];
  const float* edge_s = (const float*)d_in[3];
  const float* Wpn = (const float*)d_in[4];   const float* bpn = (const float*)d_in[5];
  const float* Wpe = (const float*)d_in[6];   const float* bpe = (const float*)d_in[7];
  // convs: base 8 + l*9 : qW qb kW kb vW vb eW skipW skipb
  const float* fc1W = (const float*)d_in[35]; const float* fc1b = (const float*)d_in[36];
  const float* fc2W = (const float*)d_in[37]; const float* fc2b = (const float*)d_in[38];
  const float* fc3W = (const float*)d_in[39]; const float* fc3b = (const float*)d_in[40];

  const int* src = eidx;
  const int* dst = eidx + EE;

  // ---- workspace layout ----
  char* ws = (char*)d_ws;
  size_t off = 0;
  auto alloc_b = [&](size_t bytes) -> void* {
    void* p = (void*)(ws + off);
    off = alignUp(off + bytes, 256);
    return p;
  };
  // f32 buffers
  float*    xA     = (float*)alloc_b((size_t)NN * 256 * 4);
  float*    xB     = (float*)alloc_b((size_t)NN * 256 * 4);
  float*    qbuf   = (float*)alloc_b((size_t)NN * 256 * 4);
  float*    kbuf   = (float*)alloc_b((size_t)NN * 256 * 4);
  float*    vbuf   = (float*)alloc_b((size_t)NN * 256 * 4);
  float*    ea     = (float*)alloc_b((size_t)EE * 128 * 4);
  float*    el     = (float*)alloc_b((size_t)EE * 256 * 4);
  float*    alpha  = (float*)alloc_b((size_t)EE * 4);
  float*    aexp   = (float*)alloc_b((size_t)EE * 4);
  unsigned* mkey   = (unsigned*)alloc_b((size_t)NN * 4);
  float*    ssum   = (float*)alloc_b((size_t)NN * 4);
  float*    pooled = (float*)alloc_b((size_t)BB * 256 * 4);
  float*    h1     = (float*)alloc_b((size_t)BB * 128 * 4);
  float*    h2     = (float*)alloc_b((size_t)BB * 64 * 4);
  // bf16 buffers
  __bf16*   xcat_b = (__bf16*)alloc_b((size_t)NN * 32 * 2);
  __bf16*   es_b   = (__bf16*)alloc_b((size_t)EE * 64 * 2);   // edge_s padded 39->64
  __bf16*   x_b    = (__bf16*)alloc_b((size_t)NN * 256 * 2);  // current x in bf16
  __bf16*   ea_b   = (__bf16*)alloc_b((size_t)EE * 128 * 2);
  __bf16*   wt_b   = (__bf16*)alloc_b((size_t)256 * 256 * 2); // transposed weight scratch

  auto cdiv = [](int a, int b) { return (a + b - 1) / b; };

  // gemm with on-the-fly weight transpose/convert into wt_b (stream-serialized)
  auto gemm = [&](const __bf16* A, const float* W, const float* bias, float* C,
                  int M, int K, int Kp, int Nout) {
    int nw = Nout * Kp;
    convert_wt_kernel<<<cdiv(nw, 256), 256, 0, stream>>>(W, wt_b, K, Kp, Nout);
    const int waves = 4;                               // 128 threads/block
    int tiles = (M >> 6) * (Nout >> 4);
    gemm_wmma_bf16<<<cdiv(tiles, waves), waves * 32, 0, stream>>>(A, wt_b, bias, C, M, Kp, Nout);
  };

  // ---- front end: concat + projections ----
  concat_convert_kernel<<<cdiv(NN * 32, 256), 256, 0, stream>>>(node_s, seq, xcat_b);
  convert_pad_kernel<<<cdiv(EE * 64, 256), 256, 0, stream>>>(edge_s, es_b, EE, 39, 64);

  gemm(xcat_b, Wpn, bpn, xA, NN, 26, 32, 128);          // x0  [N,128]
  gemm(es_b,   Wpe, bpe, ea, EE, 39, 64, 128);          // ea  [E,128]

  convert_pad_kernel<<<cdiv(NN * 128, 256), 256, 0, stream>>>(xA, x_b, NN, 128, 128);
  convert_pad_kernel<<<cdiv(EE * 128, 256), 256, 0, stream>>>(ea, ea_b, EE, 128, 128);

  // ---- 3 TransformerConv layers ----
  const int cins[3]  = {128, 128, 256};
  const int couts[3] = {128, 256, 256};
  float* x  = xA;
  float* xn = xB;
  const int eb = 256, egrid = cdiv(EE, eb);

  for (int l = 0; l < 3; ++l) {
    const int base = 8 + l * 9;
    const float* Wq = (const float*)d_in[base + 0]; const float* bq = (const float*)d_in[base + 1];
    const float* Wk = (const float*)d_in[base + 2]; const float* bk = (const float*)d_in[base + 3];
    const float* Wv = (const float*)d_in[base + 4]; const float* bv = (const float*)d_in[base + 5];
    const float* We = (const float*)d_in[base + 6];
    const float* Ws = (const float*)d_in[base + 7]; const float* bs = (const float*)d_in[base + 8];
    const int cin = cins[l], cout = couts[l];
    const float scale = 1.0f / sqrtf((float)cout);

    gemm(x_b,  Wq, bq,      qbuf, NN, cin, cin, cout);
    gemm(x_b,  Wk, bk,      kbuf, NN, cin, cin, cout);
    gemm(x_b,  Wv, bv,      vbuf, NN, cin, cin, cout);
    gemm(ea_b, We, nullptr, el,   EE, 128, 128, cout);   // edge embedding (no bias)
    gemm(x_b,  Ws, bs,      xn,   NN, cin, cin, cout);   // skip preloads the accumulator

    fill_u32<<<cdiv(NN, 256), 256, 0, stream>>>(mkey, 0u, NN);
    fill_f32<<<cdiv(NN, 256), 256, 0, stream>>>(ssum, 0.0f, NN);

    edge_alpha_kernel<<<egrid, eb, 0, stream>>>(qbuf, kbuf, el, src, dst,
                                                alpha, mkey, EE, cout, scale);
    edge_exp_kernel<<<egrid, eb, 0, stream>>>(alpha, dst, mkey, aexp, ssum, EE);
    edge_msg_kernel<<<egrid, eb, 0, stream>>>(vbuf, el, src, dst, aexp, ssum,
                                              xn, EE, cout);

    int n = NN * cout;
    lrelu_convert_kernel<<<cdiv(n, 256), 256, 0, stream>>>(xn, x_b, n);

    float* t = x; x = xn; xn = t;
  }

  // ---- pooling + MLP head ----
  pool_kernel<<<cdiv(BB * 256, 256), 256, 0, stream>>>(x, pooled);
  fc_kernel<<<cdiv(BB * 128, 256), 256, 0, stream>>>(pooled, fc1W, fc1b, h1, BB, 256, 128, 1);
  fc_kernel<<<cdiv(BB * 64, 256), 256, 0, stream>>>(h1, fc2W, fc2b, h2, BB, 128, 64, 1);
  fc_kernel<<<cdiv(BB, 256), 256, 0, stream>>>(h2, fc3W, fc3b, (float*)d_out, BB, 64, 1, 2);
}
